// M_74251394613325
// MI455X (gfx1250) — compile-verified, matrix operand order checked by reading, not run on hardware
//
#include <hip/hip_runtime.h>

typedef _Float16 v16h __attribute__((ext_vector_type(16)));
typedef float    v8f  __attribute__((ext_vector_type(8)));

// Problem constants (B=32, H=12, S=384, D=64)
constexpr int kBH    = 32 * 12;
constexpr int kS     = 384;
constexpr int kD     = 64;
constexpr int kMTILE = 64;          // query rows per workgroup
constexpr int kKH    = 192;         // K-half length (two halves cover S=384)
constexpr int kNCH   = kKH / 32;    // 32-wide WMMA K-chunks per half = 6
constexpr int kPSTR  = 208;         // prob row stride in f16 (192 data + 16 pad) = 416B = 13*32B
constexpr int kVSTR  = 208;         // V^T row stride in f16

__global__ __launch_bounds__(128) void attn_pv_wmma(
    const float* __restrict__ scores,   // [BH, S, S]
    const float* __restrict__ vmat,     // [BH, S, D]
    const float* __restrict__ mask,     // [BH, S, S]
    float* __restrict__ out)            // [BH, S, D]
{
    __shared__ alignas(32) _Float16 probS[kMTILE * kPSTR];  // P tile (f16), rows x K-half
    __shared__ alignas(32) _Float16 vtS[kD * kVSTR];        // V^T tile (f16), cols x K-half
    __shared__ float rowM[kMTILE];
    __shared__ float rowInv[kMTILE];

    const int tid  = threadIdx.x;
    const int lane = tid & 31;
    const int wave = tid >> 5;

    const int bh   = blockIdx.x / (kS / kMTILE);
    const int mblk = blockIdx.x % (kS / kMTILE);
    const int m0   = mblk * kMTILE;

    const long sBase = (long)bh * kS * kS;
    const long vBase = (long)bh * kS * kD;

    // ---- Phase 1: per-row softmax stats (max, 1/sum). Wave w owns rows 16w..16w+15.
    for (int rr = 0; rr < 16; ++rr) {
        const int lrow = wave * 16 + rr;
        const float* srow = scores + sBase + (long)(m0 + lrow) * kS;
        const float4* s4 = (const float4*)srow;
        float4 sv[3];
        #pragma unroll
        for (int j = 0; j < 3; ++j) sv[j] = s4[lane + 32 * j];   // 384 = 32 lanes * 4 * 3

        float m = sv[0].x;
        #pragma unroll
        for (int j = 0; j < 3; ++j) {
            m = fmaxf(m, sv[j].x); m = fmaxf(m, sv[j].y);
            m = fmaxf(m, sv[j].z); m = fmaxf(m, sv[j].w);
        }
        #pragma unroll
        for (int off = 16; off >= 1; off >>= 1)
            m = fmaxf(m, __shfl_xor(m, off, 32));

        float sum = 0.0f;
        #pragma unroll
        for (int j = 0; j < 3; ++j) {
            sum += __expf(sv[j].x - m); sum += __expf(sv[j].y - m);
            sum += __expf(sv[j].z - m); sum += __expf(sv[j].w - m);
        }
        #pragma unroll
        for (int off = 16; off >= 1; off >>= 1)
            sum += __shfl_xor(sum, off, 32);

        if (lane == 0) { rowM[lrow] = m; rowInv[lrow] = 1.0f / sum; }
    }

    // ---- Accumulators: wave computes a 16x64 f32 tile = 4 x (16x16) WMMA tiles.
    v8f acc[4];
    #pragma unroll
    for (int n = 0; n < 4; ++n)
        #pragma unroll
        for (int j = 0; j < 8; ++j) acc[n][j] = 0.0f;

    const int frag16 = lane & 15;   // A: M row within wave tile; B: N column within tile
    const int fragKh = lane >> 4;   // selects K 0..15 vs 16..31 within a 32-chunk

    // ---- Phase 2/3: two K-halves of 192; stage P(f16) and V^T(f16) in LDS, then WMMA.
    for (int h = 0; h < 2; ++h) {
        const int k0 = h * kKH;
        __syncthreads();            // previous half's WMMA reads done; stats visible

        // Stage V^T for this K-half: V rows k0..k0+191, all 64 columns -> vtS[col][k]
        for (int idx = tid; idx < kKH * kD; idx += 128) {
            const int r = idx >> 6;        // local K row 0..191
            const int c = idx & 63;        // D column
            vtS[c * kVSTR + r] = (_Float16)vmat[vBase + (long)(k0 + r) * kD + c];
        }

        // Stage probs for this K-half: p = exp(s - m) * inv_sum * mask  (f16)
        for (int rr = 0; rr < 16; ++rr) {
            const int lrow  = wave * 16 + rr;
            const float mx  = rowM[lrow];
            const float inv = rowInv[lrow];
            const float2* s2 = (const float2*)(scores + sBase + (long)(m0 + lrow) * kS + k0);
            const float2* k2 = (const float2*)(mask   + sBase + (long)(m0 + lrow) * kS + k0);
            #pragma unroll
            for (int j = 0; j < 3; ++j) {
                const int e = lane + 32 * j;          // 192 = 32 lanes * 2 * 3
                const float2 sv = s2[e];
                const float2 mk = k2[e];
                probS[lrow * kPSTR + 2 * e + 0] = (_Float16)(__expf(sv.x - mx) * inv * mk.x);
                probS[lrow * kPSTR + 2 * e + 1] = (_Float16)(__expf(sv.y - mx) * inv * mk.y);
            }
        }
        __syncthreads();

        // WMMA over this half: 6 K-chunks x 4 N-tiles
        #pragma unroll
        for (int kc = 0; kc < kNCH; ++kc) {
            const v16h a = *(const v16h*)&probS[(wave * 16 + frag16) * kPSTR + kc * 32 + fragKh * 16];
            #pragma unroll
            for (int n = 0; n < 4; ++n) {
                const v16h b = *(const v16h*)&vtS[(n * 16 + frag16) * kVSTR + kc * 32 + fragKh * 16];
                acc[n] = __builtin_amdgcn_wmma_f32_16x16x32_f16(
                    /*neg_a=*/false, a, /*neg_b=*/false, b,
                    /*c_mod=*/(short)0, acc[n], /*reuse_a=*/false, /*reuse_b=*/false);
            }
        }
    }

    // ---- Store: C layout (16x16 f32): VGPR j, lanes 0-15 -> M=j, N=lane; lanes 16-31 -> M=j+8.
    const int orow0 = m0 + wave * 16 + 8 * (lane >> 4);
    const int ocol  = lane & 15;
    float* obase = out + vBase;
    #pragma unroll
    for (int n = 0; n < 4; ++n)
        #pragma unroll
        for (int j = 0; j < 8; ++j)
            obase[(long)(orow0 + j) * kD + n * 16 + ocol] = acc[n][j];
}

extern "C" void kernel_launch(void* const* d_in, const int* in_sizes, int n_in,
                              void* d_out, int out_size, void* d_ws, size_t ws_size,
                              hipStream_t stream) {
    const float* scores = (const float*)d_in[0];   // x429 [B,H,S,S]
    const float* vmat   = (const float*)d_in[1];   // x419 [B,H,S,D]
    const float* mask   = (const float*)d_in[2];   // dropout_mask [B,H,S,S]
    float* out = (float*)d_out;                    // [B,H,S,D]

    dim3 grid(kBH * (kS / kMTILE));   // 384 * 6 = 2304 workgroups
    attn_pv_wmma<<<grid, 128, 0, stream>>>(scores, vmat, mask, out);
}
